// MeshGNN_30056181137887
// MI455X (gfx1250) — matrix operand order repaired and checked by reading, not compiled, not run on hardware
//
#include <hip/hip_runtime.h>

typedef __attribute__((ext_vector_type(16))) _Float16 v16h;
typedef __attribute__((ext_vector_type(8)))  _Float16 v8h;
typedef __attribute__((ext_vector_type(8)))  float    v8f;

#define IN_F   6
#define HID    128
#define HEADS  8
#define CH     16
#define OUTF   4

// ---------------------------------------------------------------------------
// utility: float atomic max via int/uint ordering trick (init with -inf)
// ---------------------------------------------------------------------------
__device__ __forceinline__ void atomicMaxFloat(float* addr, float val) {
    int iv = __float_as_int(val);
    if (val >= 0.0f) {
        atomicMax((int*)addr, iv);
    } else {
        atomicMin((unsigned int*)addr, (unsigned int)iv);
    }
}

// ---------------------------------------------------------------------------
// fill a float buffer with a constant (init mx=-inf, denom=0, agg=0)
// ---------------------------------------------------------------------------
__global__ void k_fill(float* __restrict__ p, float val, long long n) {
    long long i = (long long)blockIdx.x * blockDim.x + threadIdx.x;
    if (i < n) p[i] = val;
}

// ---------------------------------------------------------------------------
// weight swizzle: f32 B (128 x NC, row-major) -> f16 fragment-ordered Bsw.
// Fragment element j of lane l in tile (kt, ct) is
//   B[ kt*32 + (l>>4)*16 + j ][ ct*16 + (l&15) ]
// stored at Bsw[ ((kt*(NC/16)+ct)*32 + l)*16 + j ]  => each lane's fragment
// is one contiguous 32-byte chunk (2x global_load_b128 in the GEMM).
// ---------------------------------------------------------------------------
template <int NC>
__global__ void k_weight_swizzle(const float* __restrict__ wsrc,
                                 _Float16* __restrict__ dst) {
    int idx = blockIdx.x * 256 + threadIdx.x;
    if (idx >= HID * NC) return;
    int j    = idx & 15;
    int lane = (idx >> 4) & 31;
    int tile = idx >> 9;              // kt*(NC/16) + ct
    int ct   = tile % (NC / 16);
    int kt   = tile / (NC / 16);
    int col  = ct * 16 + (lane & 15);
    int k    = kt * 32 + (lane >> 4) * 16 + j;
    dst[idx] = (_Float16)wsrc[k * NC + col];
}

// ---------------------------------------------------------------------------
// input MLP: h = relu(LN(x @ w_in + b_in)) ; one row per 128-thread block
// also writes f16 mirror of h for the WMMA GEMM
// ---------------------------------------------------------------------------
__global__ void __launch_bounds__(HID)
k_input_mlp(const float* __restrict__ x, const float* __restrict__ w_in,
            const float* __restrict__ b_in, const float* __restrict__ lnw,
            const float* __restrict__ lnb, float* __restrict__ h,
            _Float16* __restrict__ hf16, int n) {
    int row = blockIdx.x;
    int c   = threadIdx.x;
    if (row >= n) return;
    float acc = b_in[c];
#pragma unroll
    for (int k = 0; k < IN_F; ++k)
        acc = fmaf(x[(long long)row * IN_F + k], w_in[k * HID + c], acc);

    __shared__ float s1[HID];
    __shared__ float s2[HID];
    s1[c] = acc; s2[c] = acc * acc;
    __syncthreads();
#pragma unroll
    for (int off = HID / 2; off > 0; off >>= 1) {
        if (c < off) { s1[c] += s1[c + off]; s2[c] += s2[c + off]; }
        __syncthreads();
    }
    float mean = s1[0] * (1.0f / HID);
    float var  = s2[0] * (1.0f / HID) - mean * mean;
    float r    = rsqrtf(var + 1e-5f);
    float y    = fmaxf((acc - mean) * r * lnw[c] + lnb[c], 0.0f);
    h[(long long)row * HID + c]    = y;
    hf16[(long long)row * HID + c] = (_Float16)y;
}

// ---------------------------------------------------------------------------
// bias + LayerNorm + ReLU (+ optional residual), fused f16 mirror output:
//   out[row] = (resid ? resid[row] : 0) + relu(LN(in[row] + bias))
// ---------------------------------------------------------------------------
__global__ void __launch_bounds__(HID)
k_bias_ln_relu(const float* __restrict__ in, float* out,
               const float* __restrict__ bias, const float* __restrict__ lnw,
               const float* __restrict__ lnb, const float* resid,
               _Float16* outf16, int n) {
    int row = blockIdx.x;
    int c   = threadIdx.x;
    if (row >= n) return;
    float a = in[(long long)row * HID + c] + bias[c];
    float res = (resid != nullptr) ? resid[(long long)row * HID + c] : 0.0f;

    __shared__ float s1[HID];
    __shared__ float s2[HID];
    s1[c] = a; s2[c] = a * a;
    __syncthreads();
#pragma unroll
    for (int off = HID / 2; off > 0; off >>= 1) {
        if (c < off) { s1[c] += s1[c + off]; s2[c] += s2[c + off]; }
        __syncthreads();
    }
    float mean = s1[0] * (1.0f / HID);
    float var  = s2[0] * (1.0f / HID) - mean * mean;
    float r    = rsqrtf(var + 1e-5f);
    float y    = res + fmaxf((a - mean) * r * lnw[c] + lnb[c], 0.0f);
    out[(long long)row * HID + c] = y;
    if (outf16 != nullptr) outf16[(long long)row * HID + c] = (_Float16)y;
}

// ---------------------------------------------------------------------------
// WMMA GEMM: C(f32, Mpad x NCOLS) = A(f16, Mpad x 128) @ Bsw(fragment-ordered)
// A and C are padded to a multiple of 64 rows (workspace), so there are NO
// row guards: unconditional loads and stores (immediate-offset b32 stores).
// Block: 256 threads = 8 waves; 64 rows per block; A staged in LDS (16 KB).
// Wave w: row-tile (w&3)*16, col-half (w>>2) -> NCOLS/32 col tiles of 16.
// K = 128 = 4 steps of v_wmma_f32_16x16x32_f16.
// ---------------------------------------------------------------------------
template <int NCOLS>
__global__ void __launch_bounds__(256)
k_gemm_wmma(const _Float16* __restrict__ A, const _Float16* __restrict__ Bsw,
            float* __restrict__ C) {
    __shared__ _Float16 As[64 * HID];      // 16 KB

    const int tid  = threadIdx.x;
    const int row0 = blockIdx.x * 64;

    // cooperative load of A tile (64 x 128), unconditional (padded buffer)
    {
        const uint4* src = (const uint4*)A + (long long)row0 * 16;
        uint4* dst = (uint4*)As;
        for (int i = tid; i < 64 * 16; i += 256) dst[i] = src[i];
    }
    __syncthreads();

    const int wave   = tid >> 5;
    const int lane   = tid & 31;
    const int m16    = lane & 15;
    const int laneHi = lane >> 4;            // 0 or 1
    const int rowT   = (wave & 3) * 16;      // row tile base within block
    const int ctBase = (wave >> 2) * (NCOLS / 32);

    // preload A fragments for all 4 K-steps
    // lane's fragment: elems 0..7 = K [kt*32+16*laneHi, +8),
    //                  elems 8..15 = K [kt*32+16*laneHi+8, +8)
    union AF { v16h v; struct { v8h lo, hi; } p; };
    v16h afrag[4];
    const int arow = rowT + m16;
    const v8h* rowp = (const v8h*)(As + arow * HID);   // 16B chunks of the row
#pragma unroll
    for (int kt = 0; kt < 4; ++kt) {
        AF a;
        a.p.lo = rowp[kt * 4 + 2 * laneHi];
        a.p.hi = rowp[kt * 4 + 2 * laneHi + 1];
        afrag[kt] = a.v;
    }

    const v16h* bf = (const v16h*)Bsw;       // one fragment per (tile, lane)

#pragma unroll
    for (int j = 0; j < NCOLS / 32; ++j) {
        const int ct = ctBase + j;
        v8f acc = {};
#pragma unroll
        for (int kt = 0; kt < 4; ++kt) {
            v16h bfrag = bf[(kt * (NCOLS / 16) + ct) * 32 + lane];
            acc = __builtin_amdgcn_wmma_f32_16x16x32_f16(
                false, afrag[kt], false, bfrag, (short)0, acc, false, false);
        }
        // D layout: col = lane&15, row = v + 8*laneHi; unconditional stores
        float* cbase = C + (long long)(row0 + rowT + laneHi * 8) * NCOLS
                         + ct * 16 + m16;
#pragma unroll
        for (int v = 0; v < 8; ++v) cbase[v * NCOLS] = acc[v];
    }
}

// ---------------------------------------------------------------------------
// attention logits per node/head: al_s = <xw, a_src>, al_d = <xw, a_dst>
// ---------------------------------------------------------------------------
__global__ void k_alpha(const float* __restrict__ xw, const float* __restrict__ a_src,
                        const float* __restrict__ a_dst, float* __restrict__ als,
                        float* __restrict__ ald, int n) {
    long long idx = (long long)blockIdx.x * blockDim.x + threadIdx.x;
    if (idx >= (long long)n * HEADS) return;
    long long node = idx >> 3;
    int hh = (int)(idx & 7);
    const float* v  = xw + node * HID + hh * CH;
    const float* as = a_src + hh * CH;
    const float* ad = a_dst + hh * CH;
    float ss = 0.f, sd = 0.f;
#pragma unroll
    for (int c = 0; c < CH; ++c) {
        float t = v[c];
        ss = fmaf(t, as[c], ss);
        sd = fmaf(t, ad[c], sd);
    }
    als[idx] = ss;
    ald[idx] = sd;
}

// ---------------------------------------------------------------------------
// edge pass 1: segment max of leaky_relu(al_s[src]+al_d[dst]) into mx[dst]
// edges 0..E-1 from edge_index; edges E..E+N-1 are self loops
// ---------------------------------------------------------------------------
__global__ void k_edge_max(const int* __restrict__ srcI, const int* __restrict__ dstI,
                           const float* __restrict__ als, const float* __restrict__ ald,
                           float* __restrict__ mx, int e, int n) {
    long long idx = (long long)blockIdx.x * blockDim.x + threadIdx.x;
    long long total = (long long)(e + n) * HEADS;
    if (idx >= total) return;
    long long ed = idx >> 3;
    int hh = (int)(idx & 7);
    int s, d;
    if (ed < e) { s = srcI[ed]; d = dstI[ed]; }
    else        { s = d = (int)(ed - e); }
    float v = als[(long long)s * HEADS + hh] + ald[(long long)d * HEADS + hh];
    v = (v > 0.f) ? v : 0.2f * v;   // leaky relu 0.2
    atomicMaxFloat(&mx[(long long)d * HEADS + hh], v);
}

// ---------------------------------------------------------------------------
// edge pass 2: denom[dst] += exp(e - mx[dst])
// ---------------------------------------------------------------------------
__global__ void k_edge_sum(const int* __restrict__ srcI, const int* __restrict__ dstI,
                           const float* __restrict__ als, const float* __restrict__ ald,
                           const float* __restrict__ mx, float* __restrict__ denom,
                           int e, int n) {
    long long idx = (long long)blockIdx.x * blockDim.x + threadIdx.x;
    long long total = (long long)(e + n) * HEADS;
    if (idx >= total) return;
    long long ed = idx >> 3;
    int hh = (int)(idx & 7);
    int s, d;
    if (ed < e) { s = srcI[ed]; d = dstI[ed]; }
    else        { s = d = (int)(ed - e); }
    float v = als[(long long)s * HEADS + hh] + ald[(long long)d * HEADS + hh];
    v = (v > 0.f) ? v : 0.2f * v;
    float p = __expf(v - mx[(long long)d * HEADS + hh]);
    atomicAdd(&denom[(long long)d * HEADS + hh], p);
}

// ---------------------------------------------------------------------------
// edge pass 3: agg[dst] += alpha * xw[src]; one wave (32 lanes) per edge,
// each lane handles 4 channels (float4 gather + 4 atomic f32 adds)
// ---------------------------------------------------------------------------
__global__ void __launch_bounds__(256)
k_edge_agg(const int* __restrict__ srcI, const int* __restrict__ dstI,
           const float* __restrict__ als, const float* __restrict__ ald,
           const float* __restrict__ mx, const float* __restrict__ denom,
           const float* __restrict__ xw, float* __restrict__ agg, int e, int n) {
    long long tid = (long long)blockIdx.x * blockDim.x + threadIdx.x;
    long long ed  = tid >> 5;
    int lane = (int)(tid & 31);
    if (ed >= (long long)(e + n)) return;
    int s, d;
    if (ed < e) { s = srcI[ed]; d = dstI[ed]; }
    else        { s = d = (int)(ed - e); }
    int hh = lane >> 2;   // 4 lanes per head, 4 channels per lane
    float v = als[(long long)s * HEADS + hh] + ald[(long long)d * HEADS + hh];
    v = (v > 0.f) ? v : 0.2f * v;
    float alpha = __expf(v - mx[(long long)d * HEADS + hh]) /
                  denom[(long long)d * HEADS + hh];

    const float* srow = xw + (long long)s * HID + lane * 4;
    __builtin_prefetch(srow, 0, 1);               // global_prefetch_b8
    float4 m = *(const float4*)srow;
    float* arow = agg + (long long)d * HID + lane * 4;
    atomicAdd(arow + 0, m.x * alpha);
    atomicAdd(arow + 1, m.y * alpha);
    atomicAdd(arow + 2, m.z * alpha);
    atomicAdd(arow + 3, m.w * alpha);
}

// ---------------------------------------------------------------------------
// elementwise: o = relu(o + b) over n rows of 64
// ---------------------------------------------------------------------------
__global__ void k_relu_bias64(float* __restrict__ o, const float* __restrict__ b,
                              long long total) {
    long long i = (long long)blockIdx.x * blockDim.x + threadIdx.x;
    if (i < total) o[i] = fmaxf(o[i] + b[i & 63], 0.0f);
}

// ---------------------------------------------------------------------------
// final projection: out(N x 4) = o2(N x 64) @ w3(64 x 4) + b3
// ---------------------------------------------------------------------------
__global__ void k_final(const float* __restrict__ o2, const float* __restrict__ w3,
                        const float* __restrict__ b3, float* __restrict__ out, int n) {
    long long idx = (long long)blockIdx.x * blockDim.x + threadIdx.x;
    if (idx >= (long long)n * OUTF) return;
    long long node = idx >> 2;
    int j = (int)(idx & 3);
    float acc = b3[j];
    const float* r = o2 + node * 64;
#pragma unroll
    for (int k = 0; k < 64; ++k) acc = fmaf(r[k], w3[k * OUTF + j], acc);
    out[idx] = acc;
}

// ---------------------------------------------------------------------------
// host launcher
// ---------------------------------------------------------------------------
extern "C" void kernel_launch(void* const* d_in, const int* in_sizes, int n_in,
                              void* d_out, int out_size, void* d_ws, size_t ws_size,
                              hipStream_t stream) {
    const float* x       = (const float*)d_in[0];
    const int*   eidx    = (const int*)  d_in[1];
    const float* w_in    = (const float*)d_in[2];
    const float* b_in    = (const float*)d_in[3];
    const float* ln_in_w = (const float*)d_in[4];
    const float* ln_in_b = (const float*)d_in[5];
    const float* Wg      = (const float*)d_in[6];
    const float* a_src   = (const float*)d_in[7];
    const float* a_dst   = (const float*)d_in[8];
    const float* bg      = (const float*)d_in[9];
    const float* lnw     = (const float*)d_in[10];
    const float* lnb     = (const float*)d_in[11];
    const float* w1      = (const float*)d_in[12];
    const float* b1      = (const float*)d_in[13];
    const float* lnow    = (const float*)d_in[14];
    const float* lnob    = (const float*)d_in[15];
    const float* w2      = (const float*)d_in[16];
    const float* b2      = (const float*)d_in[17];
    const float* w3      = (const float*)d_in[18];
    const float* b3      = (const float*)d_in[19];
    float* out = (float*)d_out;

    const int n     = in_sizes[0] / IN_F;        // 100000
    const int e     = in_sizes[1] / 2;           // 600000
    const int n_pad = (n + 63) & ~63;            // GEMM row padding
    const int* srcI = eidx;
    const int* dstI = eidx + e;

    // carve workspace (GEMM-facing buffers padded to n_pad rows)
    char* w = (char*)d_ws;
    auto carve = [&](size_t bytes) {
        char* p = w;
        w += (bytes + 255) & ~(size_t)255;
        return p;
    };
    float*    h     = (float*)   carve((size_t)n_pad * HID * 4);
    _Float16* hf16  = (_Float16*)carve((size_t)n_pad * HID * 2);
    _Float16* wf16  = (_Float16*)carve((size_t)HID * HID * 2);  // swizzled weights
    float*    xw    = (float*)   carve((size_t)n_pad * HID * 4);
    float*    als   = (float*)   carve((size_t)n * HEADS * 4);
    float*    ald   = (float*)   carve((size_t)n * HEADS * 4);
    float*    mx    = (float*)   carve((size_t)n * HEADS * 4);
    float*    denom = (float*)   carve((size_t)n * HEADS * 4);
    float*    agg   = (float*)   carve((size_t)n_pad * HID * 4); // also o2 (N x 64)

    const long long nh   = (long long)n * HID;
    const long long nhd  = (long long)n * HEADS;
    const long long edh  = (long long)(e + n) * HEADS;
    const long long edw  = (long long)(e + n) * 32;

    auto blocks = [](long long work, int bs) { return (unsigned)((work + bs - 1) / bs); };
    const int gemmGrid = n_pad / 64;

    // ---- input MLP (writes h and its f16 mirror) ----
    k_input_mlp<<<n, HID, 0, stream>>>(x, w_in, b_in, ln_in_w, ln_in_b, h, hf16, n);

    // ---- GAT layers ----
    for (int i = 0; i < 4; ++i) {
        k_weight_swizzle<HID><<<HID * HID / 256, 256, 0, stream>>>(
            Wg + (size_t)i * HID * HID, wf16);

        k_gemm_wmma<HID><<<gemmGrid, 256, 0, stream>>>(hf16, wf16, xw);

        k_alpha<<<blocks(nhd, 256), 256, 0, stream>>>(
            xw, a_src + i * HEADS * CH, a_dst + i * HEADS * CH, als, ald, n);

        k_fill<<<blocks(nhd, 256), 256, 0, stream>>>(mx, -INFINITY, nhd);
        k_fill<<<blocks(nhd, 256), 256, 0, stream>>>(denom, 0.0f, nhd);
        k_fill<<<blocks(nh, 256), 256, 0, stream>>>(agg, 0.0f, nh);

        k_edge_max<<<blocks(edh, 256), 256, 0, stream>>>(srcI, dstI, als, ald, mx, e, n);
        k_edge_sum<<<blocks(edh, 256), 256, 0, stream>>>(srcI, dstI, als, ald, mx, denom, e, n);
        k_edge_agg<<<blocks(edw, 256), 256, 0, stream>>>(srcI, dstI, als, ald, mx, denom,
                                                         xw, agg, e, n);

        // h = h + relu(LN(agg + bg)) ; refresh f16 mirror
        k_bias_ln_relu<<<n, HID, 0, stream>>>(agg, h, bg + i * HID,
                                              lnw + i * HID, lnb + i * HID, h, hf16, n);
    }

    // ---- output MLP ----
    k_weight_swizzle<HID><<<HID * HID / 256, 256, 0, stream>>>(w1, wf16);
    k_gemm_wmma<HID><<<gemmGrid, 256, 0, stream>>>(hf16, wf16, xw);
    k_bias_ln_relu<<<n, HID, 0, stream>>>(xw, xw, b1, lnow, lnob, nullptr, hf16, n);

    k_weight_swizzle<64><<<HID * 64 / 256, 256, 0, stream>>>(w2, wf16);
    k_gemm_wmma<64><<<gemmGrid, 256, 0, stream>>>(hf16, wf16, agg);   // agg = o2
    k_relu_bias64<<<blocks((long long)n * 64, 256), 256, 0, stream>>>(
        agg, b2, (long long)n * 64);

    k_final<<<blocks((long long)n * OUTF, 256), 256, 0, stream>>>(agg, w3, b3, out, n);
}
